// AdaptivePhysicallyConstrainedAttention_86268713108281
// MI455X (gfx1250) — matrix operand order repaired and checked by reading, not compiled, not run on hardware
//
#include <hip/hip_runtime.h>
#include <math.h>
#include <stdint.h>

// Problem constants (match reference)
#define B_  4
#define L_  2048
#define D_  1024
#define H_  16
#define HD_ 64
#define M_  (B_ * L_)   // 8192 rows

typedef __attribute__((ext_vector_type(16))) _Float16     v16h;
typedef __attribute__((ext_vector_type(8)))  float        v8f;
typedef __attribute__((ext_vector_type(8)))  unsigned int v8u;

static __device__ __forceinline__ v16h as_v16h(v8u w) {
    return __builtin_bit_cast(v16h, w);
}
static __device__ __forceinline__ unsigned short f2h(float f) {
    _Float16 h = (_Float16)f;
    return __builtin_bit_cast(unsigned short, h);
}

// CDNA5 async HBM->LDS copy (no VGPR round-trip, tracked by ASYNCcnt).
// ldsaddr: byte offset of LDS destination (low 32 bits of a generic LDS ptr).
static __device__ __forceinline__ void async_b128_to_lds(unsigned ldsaddr,
                                                         const void* gaddr) {
    asm volatile("global_load_async_to_lds_b128 %0, %1, off"
                 :: "v"(ldsaddr), "v"(gaddr) : "memory");
}
static __device__ __forceinline__ void wait_asynccnt0() {
    asm volatile("s_wait_asynccnt 0x0" ::: "memory");
}
static __device__ __forceinline__ unsigned lds_off(const void* p) {
    return (unsigned)(uintptr_t)p;   // LDS generic ptr: low 32 bits = LDS offset
}

// ---------------------------------------------------------------------------
// 1) fp32 -> fp16 cast (grid-stride)
// ---------------------------------------------------------------------------
__global__ void cast_f32_to_f16_kernel(const float* __restrict__ in,
                                       unsigned short* __restrict__ out, int n) {
    int i = blockIdx.x * blockDim.x + threadIdx.x;
    int stride = gridDim.x * blockDim.x;
    for (; i < n; i += stride) out[i] = f2h(in[i]);
}

// ---------------------------------------------------------------------------
// 2) top-3 of band_importance -> column bias (0 or -inf).
//    softmax is monotonic, so top-k of softmax == top-k of raw logits.
// ---------------------------------------------------------------------------
__global__ void topk_bias_kernel(const float* __restrict__ imp,
                                 float* __restrict__ bias, int n, int ksel) {
    __shared__ float sv[256];
    __shared__ int   si[256];
    __shared__ int   chosen[8];
    const int tid = threadIdx.x;
    if (tid < 8) chosen[tid] = -1;
    __syncthreads();
    for (int k = 0; k < ksel; ++k) {
        float best = -INFINITY; int bidx = 0x7fffffff;
        for (int i = tid; i < n; i += 256) {
            bool skip = false;
            for (int j = 0; j < k; ++j) if (chosen[j] == i) skip = true;
            if (skip) continue;
            float v = imp[i];
            if (v > best || (v == best && i < bidx)) { best = v; bidx = i; }
        }
        sv[tid] = best; si[tid] = bidx;
        __syncthreads();
        for (int s = 128; s > 0; s >>= 1) {
            if (tid < s) {
                if (sv[tid + s] > sv[tid] ||
                    (sv[tid + s] == sv[tid] && si[tid + s] < si[tid])) {
                    sv[tid] = sv[tid + s]; si[tid] = si[tid + s];
                }
            }
            __syncthreads();
        }
        if (tid == 0) chosen[k] = si[0];
        __syncthreads();
    }
    for (int i = tid; i < n; i += 256) {
        bool m = false;
        for (int j = 0; j < 3; ++j) if (chosen[j] == i) m = true;
        bias[i] = m ? -INFINITY : 0.0f;
    }
}

// ---------------------------------------------------------------------------
// 3) WMMA GEMM: out[m,n] = sum_k A[m,k]*W[n,k] + bias[n]   (A: MxK, W: NxK f16)
//    mode 0: store f16 (QKV path).  mode 1: +residual, store f32 (out-proj).
//    Block: 256 thr = 8 waves, 128x128 tile; wave = 32x64 (2x4 WMMA accums).
//    Double-buffered LDS filled by async HBM->LDS copies (ASYNCcnt pipeline).
// ---------------------------------------------------------------------------
__global__ __launch_bounds__(256)
void gemm_f16_wmma_kernel(const unsigned short* __restrict__ A,
                          const unsigned short* __restrict__ W,
                          const float* __restrict__ bias,
                          unsigned short* __restrict__ out_h,
                          const float* __restrict__ resid,
                          float* __restrict__ out_f,
                          int Ndim, int Kdim, int mode) {
    __shared__ unsigned short At[2][128 * 32];
    __shared__ unsigned short Bt[2][128 * 32];
    const int tid  = threadIdx.x;
    const int lane = tid & 31;
    const int wave = tid >> 5;
    const int wm = wave & 3;                 // 4 waves along M (32 rows each)
    const int wn = wave >> 2;                // 2 waves along N (64 cols each)
    const int tile_m = blockIdx.y * 128;
    const int tile_n = blockIdx.x * 128;
    const int g  = lane >> 4;                // half-wave group (ISA frag layout)
    const int lr = lane & 15;

    v8f zero = {};
    v8f acc[2][4];
    for (int i = 0; i < 2; ++i)
        for (int j = 0; j < 4; ++j) acc[i][j] = zero;

    // async stage of one 128x32 A tile + 128x32 W tile into buffer `buf`
    auto stage = [&](int buf, int kk) {
        #pragma unroll
        for (int i = 0; i < 2; ++i) {
            int lin = (tid + i * 256) * 8;       // half-element index in 4096
            int row = lin >> 5;
            int col = lin & 31;
            async_b128_to_lds(lds_off(&At[buf][row * 32 + col]),
                              &A[(size_t)(tile_m + row) * Kdim + kk + col]);
            async_b128_to_lds(lds_off(&Bt[buf][row * 32 + col]),
                              &W[(size_t)(tile_n + row) * Kdim + kk + col]);
        }
    };

    int cur = 0;
    stage(0, 0);
    for (int kk = 0; kk < Kdim; kk += 32) {
        wait_asynccnt0();          // our async fills for buf `cur` done
        __syncthreads();           // everyone's fills done / prev reads done
        if (kk + 32 < Kdim) stage(cur ^ 1, kk + 32);
        if (kk + 64 < Kdim) {      // warm L2 two tiles ahead
            __builtin_prefetch(&A[(size_t)(tile_m + (tid >> 1)) * Kdim + kk + 64], 0, 1);
            __builtin_prefetch(&W[(size_t)(tile_n + (tid >> 1)) * Kdim + kk + 64], 0, 1);
        }
        // A fragments (16x32, row = lane%16, K packed per ISA 7.12.2)
        v16h af[2];
        #pragma unroll
        for (int fm = 0; fm < 2; ++fm) {
            const unsigned short* p = &At[cur][(wm * 32 + fm * 16 + lr) * 32];
            v8u w;
            #pragma unroll
            for (int v = 0; v < 4; ++v) {
                w[v]     = *(const unsigned int*)&p[g * 8 + 2 * v];
                w[v + 4] = *(const unsigned int*)&p[16 + g * 8 + 2 * v];
            }
            af[fm] = as_v16h(w);
        }
        #pragma unroll
        for (int fn = 0; fn < 4; ++fn) {
            // B fragment (32x16): col = lane%16, K packed like A (g*8 + 2v, 16 + g*8 + 2v)
            const unsigned short* p = &Bt[cur][(wn * 64 + fn * 16 + lr) * 32];
            v8u w;
            #pragma unroll
            for (int v = 0; v < 4; ++v) {
                w[v]     = *(const unsigned int*)&p[g * 8 + 2 * v];
                w[v + 4] = *(const unsigned int*)&p[16 + g * 8 + 2 * v];
            }
            v16h bf = as_v16h(w);
            #pragma unroll
            for (int fm = 0; fm < 2; ++fm)
                acc[fm][fn] = __builtin_amdgcn_wmma_f32_16x16x32_f16(
                    false, af[fm], false, bf, (short)0, acc[fm][fn], false, false);
        }
        cur ^= 1;
    }
    // epilogue (C layout: row = v + 8*g, col = lane%16)
    for (int fm = 0; fm < 2; ++fm) {
        for (int fn = 0; fn < 4; ++fn) {
            int colg = tile_n + wn * 64 + fn * 16 + lr;
            float bb = bias[colg];
            #pragma unroll
            for (int v = 0; v < 8; ++v) {
                int rowg = tile_m + wm * 32 + fm * 16 + g * 8 + v;
                size_t off = (size_t)rowg * Ndim + colg;
                float val = acc[fm][fn][v] + bb;
                if (mode == 0) out_h[off] = f2h(val);
                else           out_f[off] = val + resid[off];
            }
        }
    }
}

// ---------------------------------------------------------------------------
// 4) Flash attention: block = 4 waves, each wave owns 16 q-rows.
//    K/V double-buffered in LDS via async copies (32 keys x 64 dims each),
//    online softmax in f32, scores & PV through v_wmma_f32_16x16x32_f16.
// ---------------------------------------------------------------------------
__global__ __launch_bounds__(128)
void attention_kernel(const unsigned short* __restrict__ Q,
                      const unsigned short* __restrict__ Km,
                      const unsigned short* __restrict__ V,
                      const float* __restrict__ bias,
                      unsigned short* __restrict__ Ctx) {
    __shared__ unsigned short Ksh[2][32 * 64];
    __shared__ unsigned short Vsh[2][32 * 64];
    __shared__ float          Ssh[4][16 * 32];
    __shared__ unsigned short Psh[4][16 * 32];
    __shared__ float          stat[4][16];

    const int tid  = threadIdx.x;
    const int lane = tid & 31;
    const int wave = tid >> 5;
    const int g  = lane >> 4;
    const int lr = lane & 15;
    const int b = blockIdx.z, h = blockIdx.y, qt = blockIdx.x;
    const int qrow0 = b * L_ + qt * 64 + wave * 16;   // row into (8192,1024)
    const int hc = h * HD_;

    auto stage_kv = [&](int buf, int kb) {
        #pragma unroll
        for (int i = 0; i < 2; ++i) {
            int lin = (tid + i * 128) * 8;
            int row = lin >> 6, col = lin & 63;
            size_t goff = (size_t)(b * L_ + kb + row) * D_ + hc + col;
            async_b128_to_lds(lds_off(&Ksh[buf][lin]), &Km[goff]);
            async_b128_to_lds(lds_off(&Vsh[buf][lin]), &V[goff]);
        }
    };

    // Q A-fragments: loop-invariant, 2 chunks over hd=64
    v16h aQ[2];
    {
        const unsigned short* p = &Q[(size_t)(qrow0 + lr) * D_ + hc];
        for (int c = 0; c < 2; ++c) {
            v8u w;
            for (int v = 0; v < 4; ++v) {
                w[v]     = *(const unsigned int*)&p[c * 32 + g * 8 + 2 * v];
                w[v + 4] = *(const unsigned int*)&p[c * 32 + 16 + g * 8 + 2 * v];
            }
            aQ[c] = as_v16h(w);
        }
    }

    v8f zero = {};
    v8f oacc[4];
    for (int i = 0; i < 4; ++i) oacc[i] = zero;
    float m_run = -INFINITY, l_run = 0.0f;

    int cur = 0;
    stage_kv(0, 0);
    for (int kb = 0; kb < L_; kb += 32) {
        wait_asynccnt0();
        __syncthreads();           // K/V chunk `cur` resident for all waves
        if (kb + 32 < L_) stage_kv(cur ^ 1, kb + 32);

        // scores S = Q K^T  (2 key tiles x 2 hd chunks)
        v8f sacc[2]; sacc[0] = zero; sacc[1] = zero;
        #pragma unroll
        for (int fn = 0; fn < 2; ++fn) {
            int keyl = fn * 16 + lr;
            #pragma unroll
            for (int c = 0; c < 2; ++c) {
                const unsigned short* p = &Ksh[cur][keyl * 64 + c * 32];
                v8u w;
                #pragma unroll
                for (int v = 0; v < 4; ++v) {
                    w[v]     = *(const unsigned int*)&p[g * 8 + 2 * v];
                    w[v + 4] = *(const unsigned int*)&p[16 + g * 8 + 2 * v];
                }
                sacc[fn] = __builtin_amdgcn_wmma_f32_16x16x32_f16(
                    false, aQ[c], false, as_v16h(w), (short)0, sacc[fn], false, false);
            }
        }
        // scale + column bias, spill to LDS in (row, col) order
        #pragma unroll
        for (int fn = 0; fn < 2; ++fn) {
            float bb = bias[kb + fn * 16 + lr];
            #pragma unroll
            for (int v = 0; v < 8; ++v)
                Ssh[wave][(g * 8 + v) * 32 + fn * 16 + lr] = sacc[fn][v] * 0.125f + bb;
        }
        __syncthreads();

        // online softmax: lane handles row lr, cols g*16..g*16+15
        {
            float* Sr = &Ssh[wave][lr * 32 + g * 16];
            float lmax = -INFINITY;
            #pragma unroll
            for (int c = 0; c < 16; ++c) lmax = fmaxf(lmax, Sr[c]);
            float omax  = fmaxf(lmax, __shfl_xor(lmax, 16, 32));
            float m_new = fmaxf(m_run, omax);
            float alpha = __expf(m_run - m_new);
            float lsum = 0.0f;
            unsigned short* Pr = &Psh[wave][lr * 32 + g * 16];
            #pragma unroll
            for (int c = 0; c < 16; ++c) {
                float pv = __expf(Sr[c] - m_new);
                Pr[c] = f2h(pv);
                lsum += pv;
            }
            float rsum = lsum + __shfl_xor(lsum, 16, 32);
            l_run = l_run * alpha + rsum;
            m_run = m_new;
            if (g == 0) stat[wave][lr] = alpha;
        }
        __syncthreads();

        // rescale running output accumulators by per-row alpha
        #pragma unroll
        for (int v = 0; v < 8; ++v) {
            float a = stat[wave][g * 8 + v];
            #pragma unroll
            for (int fn = 0; fn < 4; ++fn) oacc[fn][v] *= a;
        }

        // P fragment (16x32 A-layout) from LDS
        v16h aP;
        {
            const unsigned short* p = &Psh[wave][lr * 32];
            v8u w;
            #pragma unroll
            for (int v = 0; v < 4; ++v) {
                w[v]     = *(const unsigned int*)&p[g * 8 + 2 * v];
                w[v + 4] = *(const unsigned int*)&p[16 + g * 8 + 2 * v];
            }
            aP = as_v16h(w);
        }
        // O += P V   (4 dim tiles of 16)
        #pragma unroll
        for (int fn = 0; fn < 4; ++fn) {
            int dim = fn * 16 + lr;
            v8u w;
            #pragma unroll
            for (int v = 0; v < 8; ++v) {
                int kr = (v < 4) ? (g * 8 + 2 * v) : (16 + g * 8 + 2 * (v - 4));
                unsigned int lo = Vsh[cur][kr       * 64 + dim];
                unsigned int hi = Vsh[cur][(kr + 1) * 64 + dim];
                w[v] = lo | (hi << 16);
            }
            oacc[fn] = __builtin_amdgcn_wmma_f32_16x16x32_f16(
                false, aP, false, as_v16h(w), (short)0, oacc[fn], false, false);
        }
        __syncthreads();
        cur ^= 1;
    }

    if (g == 0) stat[wave][lr] = 1.0f / l_run;
    __syncthreads();
    #pragma unroll
    for (int fn = 0; fn < 4; ++fn) {
        int colg = hc + fn * 16 + lr;
        #pragma unroll
        for (int v = 0; v < 8; ++v) {
            int rowg = qrow0 + g * 8 + v;
            Ctx[(size_t)rowg * D_ + colg] = f2h(oacc[fn][v] * stat[wave][g * 8 + v]);
        }
    }
}

// ---------------------------------------------------------------------------
// 5) Row LayerNorm: one 256-thread block per row (4 floats per thread).
// ---------------------------------------------------------------------------
__global__ __launch_bounds__(256)
void layernorm_kernel(const float* __restrict__ Y,
                      const float* __restrict__ gamma,
                      const float* __restrict__ beta,
                      float* __restrict__ out) {
    __shared__ float red[256];
    const int row = blockIdx.x;
    const int tid = threadIdx.x;
    const float4 v = *(const float4*)&Y[(size_t)row * D_ + tid * 4];
    red[tid] = v.x + v.y + v.z + v.w;
    __syncthreads();
    for (int s = 128; s > 0; s >>= 1) { if (tid < s) red[tid] += red[tid + s]; __syncthreads(); }
    float mu = red[0] * (1.0f / D_);
    __syncthreads();
    float dx = v.x - mu, dy = v.y - mu, dz = v.z - mu, dw = v.w - mu;
    red[tid] = dx * dx + dy * dy + dz * dz + dw * dw;
    __syncthreads();
    for (int s = 128; s > 0; s >>= 1) { if (tid < s) red[tid] += red[tid + s]; __syncthreads(); }
    float inv = rsqrtf(red[0] * (1.0f / D_) + 1e-5f);
    const float4 gm = *(const float4*)&gamma[tid * 4];
    const float4 bt = *(const float4*)&beta[tid * 4];
    float4 o;
    o.x = dx * inv * gm.x + bt.x;
    o.y = dy * inv * gm.y + bt.y;
    o.z = dz * inv * gm.z + bt.z;
    o.w = dw * inv * gm.w + bt.w;
    *(float4*)&out[(size_t)row * D_ + tid * 4] = o;
}

// ---------------------------------------------------------------------------
// launch
// ---------------------------------------------------------------------------
extern "C" void kernel_launch(void* const* d_in, const int* in_sizes, int n_in,
                              void* d_out, int out_size, void* d_ws, size_t ws_size,
                              hipStream_t stream) {
    (void)in_sizes; (void)n_in; (void)out_size; (void)ws_size;
    const float* x     = (const float*)d_in[0];
    const float* bimp  = (const float*)d_in[1];
    const float* Wq    = (const float*)d_in[2];
    const float* bq    = (const float*)d_in[3];
    const float* Wk    = (const float*)d_in[4];
    const float* bk    = (const float*)d_in[5];
    const float* Wv    = (const float*)d_in[6];
    const float* bv    = (const float*)d_in[7];
    const float* Wo    = (const float*)d_in[8];
    const float* bo    = (const float*)d_in[9];
    const float* gamma = (const float*)d_in[10];
    const float* beta  = (const float*)d_in[11];
    float* out = (float*)d_out;

    char* ws = (char*)d_ws;
    size_t off = 0;
    auto alloc = [&](size_t bytes) -> void* {
        void* p = ws + off;
        off += (bytes + 255) & ~(size_t)255;
        return p;
    };
    unsigned short* xb   = (unsigned short*)alloc((size_t)M_ * D_ * 2);
    unsigned short* Wqb  = (unsigned short*)alloc((size_t)D_ * D_ * 2);
    unsigned short* Wkb  = (unsigned short*)alloc((size_t)D_ * D_ * 2);
    unsigned short* Wvb  = (unsigned short*)alloc((size_t)D_ * D_ * 2);
    unsigned short* Wob  = (unsigned short*)alloc((size_t)D_ * D_ * 2);
    unsigned short* Qb   = (unsigned short*)alloc((size_t)M_ * D_ * 2);
    unsigned short* Kb   = (unsigned short*)alloc((size_t)M_ * D_ * 2);
    unsigned short* Vb   = (unsigned short*)alloc((size_t)M_ * D_ * 2);
    unsigned short* Ctxb = (unsigned short*)alloc((size_t)M_ * D_ * 2);
    float*          Yf   = (float*)alloc((size_t)M_ * D_ * 4);
    float*          cbias= (float*)alloc((size_t)L_ * 4);

    cast_f32_to_f16_kernel<<<1024, 256, 0, stream>>>(x, xb, M_ * D_);
    cast_f32_to_f16_kernel<<<512, 256, 0, stream>>>(Wq, Wqb, D_ * D_);
    cast_f32_to_f16_kernel<<<512, 256, 0, stream>>>(Wk, Wkb, D_ * D_);
    cast_f32_to_f16_kernel<<<512, 256, 0, stream>>>(Wv, Wvb, D_ * D_);
    cast_f32_to_f16_kernel<<<512, 256, 0, stream>>>(Wo, Wob, D_ * D_);
    topk_bias_kernel<<<1, 256, 0, stream>>>(bimp, cbias, L_, 3);

    dim3 ggrid(D_ / 128, M_ / 128);
    gemm_f16_wmma_kernel<<<ggrid, 256, 0, stream>>>(xb, Wqb, bq, Qb, nullptr, nullptr, D_, D_, 0);
    gemm_f16_wmma_kernel<<<ggrid, 256, 0, stream>>>(xb, Wkb, bk, Kb, nullptr, nullptr, D_, D_, 0);
    gemm_f16_wmma_kernel<<<ggrid, 256, 0, stream>>>(xb, Wvb, bv, Vb, nullptr, nullptr, D_, D_, 0);

    attention_kernel<<<dim3(L_ / 64, H_, B_), 128, 0, stream>>>(Qb, Kb, Vb, cbias, Ctxb);

    gemm_f16_wmma_kernel<<<ggrid, 256, 0, stream>>>(Ctxb, Wob, bo, nullptr, x, Yf, D_, D_, 1);
    layernorm_kernel<<<M_, 256, 0, stream>>>(Yf, gamma, beta, out);
}
